// MultWaveGCUNetWithoutDWT_18640158064670
// MI455X (gfx1250) — matrix operand order repaired and checked by reading, not compiled, not run on hardware
//
#include <hip/hip_runtime.h>
#include <hip/hip_bf16.h>

typedef __attribute__((ext_vector_type(16))) _Float16 v16h;
typedef __attribute__((ext_vector_type(8)))  float    v8f;

#define B_   8
#define T_   1024
#define N_   300
#define NP   320      // N padded to multiple of 16
#define D_   64
#define C_   32
#define KS   10       // NP / 32 K-steps
#define MT   20       // NP / 16 v-tiles
#define TT   64       // T  / 16 t-tiles
#define TOPK 30
#define NEG  0.01f

// 16-bit A-matrix (16x32) lane layout (contiguous k, matching B fragments):
// lane L (g = L/16, m = L%16), packed half idx:
// K = g*16 + idx
__device__ __forceinline__ int kmapA(int g, int idx) {
    return g * 16 + idx;
}

__device__ __forceinline__ float leaky(float v) { return v >= 0.f ? v : NEG * v; }

// ---------------- adjacency construction (small, VALU) ----------------

__global__ void k_embed(const float* __restrict__ e1, const float* __restrict__ e2,
                        const float* __restrict__ w1, const float* __restrict__ b1,
                        const float* __restrict__ w2, const float* __restrict__ b2,
                        float* __restrict__ n1, float* __restrict__ n2) {
    int tid = blockIdx.x * 256 + threadIdx.x;
    if (tid >= N_ * D_) return;
    int i = tid / D_, d = tid % D_;
    float a1 = b1[d], a2 = b2[d];
    for (int e = 0; e < D_; ++e) {
        a1 += e1[i * D_ + e] * w1[d * D_ + e];
        a2 += e2[i * D_ + e] * w2[d * D_ + e];
    }
    n1[tid] = tanhf(3.0f * a1);
    n2[tid] = tanhf(3.0f * a2);
}

__global__ void k_adj(const float* __restrict__ n1, const float* __restrict__ n2,
                      float* __restrict__ araw) {
    int tid = blockIdx.x * 256 + threadIdx.x;
    if (tid >= N_ * N_) return;
    int i = tid / N_, j = tid % N_;
    float s = 0.f;
    for (int e = 0; e < D_; ++e)
        s += n1[i * D_ + e] * n2[j * D_ + e] - n2[i * D_ + e] * n1[j * D_ + e];
    float v = tanhf(3.0f * s);
    araw[tid] = v > 0.f ? v : 0.f;
}

// top-k by rank count (ties broken by lower index, matching lax.top_k)
__global__ void k_topk(const float* __restrict__ araw, float* __restrict__ akept) {
    int tid = blockIdx.x * 256 + threadIdx.x;
    if (tid >= N_ * N_) return;
    int i = tid / N_, j = tid % N_;
    float v = araw[i * N_ + j];
    int rank = 0;
    for (int l = 0; l < N_; ++l) {
        float u = araw[i * N_ + l];
        rank += (u > v) || (u == v && l < j);
    }
    akept[tid] = (rank < TOPK) ? v : 0.f;
}

__global__ void k_rowsum(const float* __restrict__ akept, float* __restrict__ rs) {
    int i = blockIdx.x * 256 + threadIdx.x;
    if (i >= N_) return;
    float s = 1.0f;                       // self loop
    for (int j = 0; j < N_; ++j) s += akept[i * N_ + j];
    rs[i] = s;
}

// pack row-normalized adjacency (+I) into WMMA A-fragments, f16, zero padded
__global__ void k_pack_an(const float* __restrict__ akept, const float* __restrict__ rs,
                          _Float16* __restrict__ anA) {
    int tid = blockIdx.x * 256 + threadIdx.x;
    if (tid >= MT * KS * 32) return;
    int lane = tid & 31;
    int ks = (tid >> 5) % KS;
    int mt = tid / (32 * KS);
    int g = lane >> 4;
    int vrow = mt * 16 + (lane & 15);
    v16h out;
#pragma unroll
    for (int idx = 0; idx < 16; ++idx) {
        int w = ks * 32 + kmapA(g, idx);
        float val = 0.f;
        if (vrow < N_ && w < N_) {
            float a = akept[vrow * N_ + w] + (vrow == w ? 1.f : 0.f);
            val = a / rs[vrow];
        }
        out[idx] = (_Float16)val;
    }
    *(v16h*)(anA + (size_t)tid * 16) = out;
}

// pack W1eff / W2eff (mixprop folded into the 1x1 mix) as WMMA A-fragments
__global__ void k_pack_w(const float* __restrict__ mlpw, _Float16* __restrict__ wP) {
    int tid = threadIdx.x;                 // 128 threads: (which, mt, lane)
    if (tid >= 4 * 32) return;
    int lane = tid & 31;
    int mt = (tid >> 5) & 1;
    int which = tid >> 6;
    int g = lane >> 4;
    int m = mt * 16 + (lane & 15);         // output channel
    v16h out;
#pragma unroll
    for (int idx = 0; idx < 16; ++idx) {
        int k = kmapA(g, idx);
        float v = (which == 0)
                      ? (mlpw[m * 64 + k] + 0.05f * mlpw[m * 64 + 32 + k])
                      : (0.95f * mlpw[m * 64 + 32 + k]);
        out[idx] = (_Float16)v;
    }
    *(v16h*)(wP + (size_t)tid * 16) = out;
}

// ---------------- pack h = leaky(w_c*x + b_c) into WMMA B-fragments ----------------
// layout: hB[((((b*C + c)*KS + ks)*TT + tt)*32 + lane)*16 + idx]
// B-matrix (32x16) lane layout: tcol = lane%16, k = (lane/16)*16 + idx

__global__ void k_pack_h(const float* __restrict__ x, const float* __restrict__ cw,
                         const float* __restrict__ cb, _Float16* __restrict__ hB) {
    size_t tid = (size_t)blockIdx.x * 256 + threadIdx.x;   // B*C*KS*TT*32 threads
    int lane = (int)(tid & 31);
    size_t r = tid >> 5;
    int tt = (int)(r & 63); r >>= 6;
    int ks = (int)(r % KS); r /= KS;
    int c = (int)(r & 31); r >>= 5;
    int b = (int)r;
    if (b >= B_) return;
    int g = lane >> 4, tl = lane & 15;
    int t = tt * 16 + tl;
    int w0 = ks * 32 + g * 16;
    float wc = cw[c], bc = cb[c];
    const float* xp = x + ((size_t)b * T_ + t) * N_;
    v16h out;
#pragma unroll
    for (int i = 0; i < 16; ++i) {
        int w = w0 + i;
        float xv = (w < N_) ? xp[w] : 0.f;
        out[i] = (_Float16)leaky(wc * xv + bc);
    }
    *(v16h*)(hB + tid * 16) = out;
}

// ---------------- fused propagation GEMM + folded-mixprop MLP ----------------
// one wave per (b, v-tile, t-tile). Propagation p = AN @ H via 10 wmma per
// channel, p tile staged to LDS as f16, then MLP: latent = W1eff@h + W2eff@p.

__global__ __launch_bounds__(32) void k_prop_mlp(
    const _Float16* __restrict__ anA, const _Float16* __restrict__ hB,
    const _Float16* __restrict__ wP, const float* __restrict__ x,
    const float* __restrict__ cw, const float* __restrict__ cb,
    const float* __restrict__ mlpb, _Float16* __restrict__ latent) {
    __shared__ _Float16 pls[C_ * 16 * 16];   // p tile: [channel][vlocal][tlocal]

    const int lane = threadIdx.x;
    const int tt = blockIdx.x, vt = blockIdx.y, b = blockIdx.z;
    const int g = lane >> 4, tl = lane & 15;

    // adjacency A-fragments: reused across all 32 channels
    v16h afr[KS];
#pragma unroll
    for (int ks = 0; ks < KS; ++ks)
        afr[ks] = *(const v16h*)(anA + ((size_t)(vt * KS + ks) * 32 + lane) * 16);

    for (int c = 0; c < C_; ++c) {
        v8f acc = {};
#pragma unroll
        for (int ks = 0; ks < KS; ++ks) {
            v16h bfr = *(const v16h*)(
                hB + ((size_t)(((b * C_ + c) * KS + ks) * TT + tt) * 32 + lane) * 16);
            acc = __builtin_amdgcn_wmma_f32_16x16x32_f16(
                false, afr[ks], false, bfr, (short)0, acc, false, false);
        }
        // D layout: tcol = lane%16, m = r + 8*(lane/16)
#pragma unroll
        for (int r = 0; r < 8; ++r) {
            int vl = r + 8 * g;
            pls[(c * 16 + vl) * 16 + tl] = (_Float16)acc[r];
        }
    }

    // MLP A-fragments (W1eff, W2eff for the two 16-row cout tiles)
    v16h w1f[2], w2f[2];
#pragma unroll
    for (int mt2 = 0; mt2 < 2; ++mt2) {
        w1f[mt2] = *(const v16h*)(wP + ((size_t)(0 * 2 + mt2) * 32 + lane) * 16);
        w2f[mt2] = *(const v16h*)(wP + ((size_t)(1 * 2 + mt2) * 32 + lane) * 16);
    }
    float cwv[16], cbv[16], mbv[2][8];
#pragma unroll
    for (int i = 0; i < 16; ++i) { cwv[i] = cw[g * 16 + i]; cbv[i] = cb[g * 16 + i]; }
#pragma unroll
    for (int mt2 = 0; mt2 < 2; ++mt2)
#pragma unroll
        for (int r = 0; r < 8; ++r) mbv[mt2][r] = mlpb[mt2 * 16 + r + 8 * g];

    const int t = tt * 16 + tl;
    for (int vl = 0; vl < 16; ++vl) {
        const int n = vt * 16 + vl;
        // h B-fragment recomputed from a single x scalar per lane
        float xv = (n < N_) ? x[((size_t)b * T_ + t) * N_ + n] : 0.f;
        v16h bh, bp;
#pragma unroll
        for (int i = 0; i < 16; ++i) {
            bh[i] = (_Float16)leaky(cwv[i] * xv + cbv[i]);
            bp[i] = pls[((g * 16 + i) * 16 + vl) * 16 + tl];
        }
#pragma unroll
        for (int mt2 = 0; mt2 < 2; ++mt2) {
            v8f acc = {};
            acc = __builtin_amdgcn_wmma_f32_16x16x32_f16(
                false, w1f[mt2], false, bh, (short)0, acc, false, false);
            acc = __builtin_amdgcn_wmma_f32_16x16x32_f16(
                false, w2f[mt2], false, bp, (short)0, acc, false, false);
#pragma unroll
            for (int r = 0; r < 8; ++r) {
                int cout = mt2 * 16 + r + 8 * g;
                float lv = acc[r] + mbv[mt2][r];
                latent[(((size_t)b * C_ + cout) * NP + n) * T_ + t] = (_Float16)lv;
            }
        }
    }
}

// ---------------- 3x3 conv (C->1) + leaky, transpose to [B,T,N] ----------------

__global__ void k_conv_out(const _Float16* __restrict__ latent,
                           const float* __restrict__ cow, const float* __restrict__ cob,
                           float* __restrict__ out) {
    size_t tid = (size_t)blockIdx.x * 256 + threadIdx.x;
    if (tid >= (size_t)B_ * T_ * N_) return;
    int n = (int)(tid % N_);
    size_t r = tid / N_;
    int t = (int)(r % T_);
    int b = (int)(r / T_);
    float acc = cob[0];
    for (int c = 0; c < C_; ++c) {
        const _Float16* lp = latent + ((size_t)b * C_ + c) * NP * T_;
#pragma unroll
        for (int dn = -1; dn <= 1; ++dn) {
            int nn = n + dn;
            if (nn < 0 || nn >= N_) continue;
#pragma unroll
            for (int dt = -1; dt <= 1; ++dt) {
                int t2 = t + dt;
                if (t2 < 0 || t2 >= T_) continue;
                acc += (float)lp[(size_t)nn * T_ + t2] * cow[c * 9 + (dn + 1) * 3 + (dt + 1)];
            }
        }
    }
    out[tid] = leaky(acc);   // out[b][t][n] == flat tid
}

// ---------------- launch ----------------

extern "C" void kernel_launch(void* const* d_in, const int* in_sizes, int n_in,
                              void* d_out, int out_size, void* d_ws, size_t ws_size,
                              hipStream_t stream) {
    const float* x       = (const float*)d_in[0];
    const float* emb1    = (const float*)d_in[1];
    const float* emb2    = (const float*)d_in[2];
    const float* lin1_w  = (const float*)d_in[3];
    const float* lin1_b  = (const float*)d_in[4];
    const float* lin2_w  = (const float*)d_in[5];
    const float* lin2_b  = (const float*)d_in[6];
    const float* cin_w   = (const float*)d_in[7];
    const float* cin_b   = (const float*)d_in[8];
    const float* mlp_w   = (const float*)d_in[9];
    const float* mlp_b   = (const float*)d_in[10];
    const float* cout_w  = (const float*)d_in[11];
    const float* cout_b  = (const float*)d_in[12];
    float* out = (float*)d_out;

    char* ws = (char*)d_ws;
    size_t off = 0;
    auto carve = [&](size_t bytes) -> char* {
        off = (off + 255) & ~(size_t)255;
        char* p = ws + off;
        off += bytes;
        return p;
    };

    float*    n1    = (float*)carve((size_t)N_ * D_ * 4);
    float*    n2    = (float*)carve((size_t)N_ * D_ * 4);
    float*    araw  = (float*)carve((size_t)N_ * N_ * 4);
    float*    akept = (float*)carve((size_t)N_ * N_ * 4);
    float*    rs    = (float*)carve((size_t)N_ * 4);
    _Float16* anA   = (_Float16*)carve((size_t)MT * KS * 32 * 16 * 2);
    _Float16* wP    = (_Float16*)carve((size_t)4 * 32 * 16 * 2);
    _Float16* hB    = (_Float16*)carve((size_t)B_ * C_ * KS * TT * 32 * 16 * 2);
    _Float16* lat   = (_Float16*)carve((size_t)B_ * C_ * NP * T_ * 2);

    k_embed<<<(N_ * D_ + 255) / 256, 256, 0, stream>>>(emb1, emb2, lin1_w, lin1_b,
                                                       lin2_w, lin2_b, n1, n2);
    k_adj<<<(N_ * N_ + 255) / 256, 256, 0, stream>>>(n1, n2, araw);
    k_topk<<<(N_ * N_ + 255) / 256, 256, 0, stream>>>(araw, akept);
    k_rowsum<<<(N_ + 255) / 256, 256, 0, stream>>>(akept, rs);
    k_pack_an<<<(MT * KS * 32 + 255) / 256, 256, 0, stream>>>(akept, rs, anA);
    k_pack_w<<<1, 128, 0, stream>>>(mlp_w, wP);

    size_t hthreads = (size_t)B_ * C_ * KS * TT * 32;
    k_pack_h<<<(unsigned)((hthreads + 255) / 256), 256, 0, stream>>>(x, cin_w, cin_b, hB);

    dim3 grid(TT, MT, B_);
    k_prop_mlp<<<grid, 32, 0, stream>>>(anA, hB, wP, x, cin_w, cin_b, mlp_b, lat);

    size_t othreads = (size_t)B_ * T_ * N_;
    k_conv_out<<<(unsigned)((othreads + 255) / 256), 256, 0, stream>>>(lat, cout_w, cout_b, out);
}
